// InheritanceLoss_52802327937232
// MI455X (gfx1250) — matrix operand order repaired and checked by reading, not compiled, not run on hardware
//
#include <hip/hip_runtime.h>
#include <hip/hip_bf16.h>

// ---------------------------------------------------------------------------
// InheritanceLoss for MI455X (gfx1250, wave32).
// Gaussian smoothing = banded-matrix GEMM on the WMMA units:
//   Out(16x16) = Tile(16x32) x Band(32x16)   (v_wmma_f32_16x16x32_f16)
// Horizontal pass stages its input tile via the Tensor Data Mover
// (tensor_load_to_lds + s_wait_tensorcnt); vertical pass uses the CDNA5
// async-to-LDS path (global_load_async_to_lds_b128 / s_wait_asynccnt).
// Band-matrix WMMA fragments are lane-only constants precomputed once.
// ---------------------------------------------------------------------------

typedef __attribute__((ext_vector_type(16))) _Float16 v16h;
typedef __attribute__((ext_vector_type(8)))  float    v8f;
typedef __attribute__((ext_vector_type(4)))  unsigned int u32x4;
typedef __attribute__((ext_vector_type(8)))  int      i32x8;
typedef __attribute__((ext_vector_type(4)))  int      i32x4;

// 15-tap Gaussian, sigma=4, normalized (precomputed; symmetric).
__constant__ float GK[15] = {
    0.0229492f, 0.0344505f, 0.0485832f, 0.0643623f, 0.0801002f,
    0.0936467f, 0.1028508f, 0.1061153f, 0.1028508f, 0.0936467f,
    0.0801002f, 0.0643623f, 0.0485832f, 0.0344505f, 0.0229492f};

// ---------------------------------------------------------------------------
// Precompute per-lane WMMA band fragments (lane-only -> 32x16 f16 tables).
//   bh: horizontal-pass B fragment  B[K][n] = g[K-n-1]       (K = e+16*half)
//   av: vertical-pass   A fragment  A[m][K] = 100*g[K-m-1]   (K = (e<8?e:e+8)+8*half)
// ---------------------------------------------------------------------------
__global__ void band_init_k(_Float16* __restrict__ bh, _Float16* __restrict__ av) {
  const int lane = threadIdx.x & 31;
  const int half = lane >> 4, m = lane & 15;
#pragma unroll
  for (int e = 0; e < 16; ++e) {
    const int Kb = e + 16 * half;
    const int jb = Kb - m - 1;
    bh[lane * 16 + e] = (_Float16)((jb >= 0 && jb < 15) ? GK[jb] : 0.0f);
    const int Ka = (e < 8 ? e : e + 8) + 8 * half;
    const int ja = Ka - m - 1;
    av[lane * 16 + e] = (_Float16)((ja >= 0 && ja < 15) ? 100.0f * GK[ja] : 0.0f);
  }
}

// ---------------------------------------------------------------------------
// Horizontal pass. Block = 8 waves = 8 adjacent 16x16 tiles (16 rows x 128
// cols output). Input tile (16 rows x 144 cols, left halo 8 / right halo 8)
// is DMA'd into LDS by the Tensor Data Mover. D# uses pad_enable (1 DWORD
// every 16) -> 153-float LDS row pitch, conflict-free transposed reads
// (153 mod 64 = 25, coprime with banks). Right/bottom OOB zero-filled by
// tensor_dim0/1; left edge handled by a branchless predicate on readback.
// ---------------------------------------------------------------------------
__global__ void smooth_h_k(const float* __restrict__ src, float* __restrict__ dst,
                           const _Float16* __restrict__ bh, int B, int H, int W) {
  __shared__ float smem[16 * 153];
  const int tid = threadIdx.x;
  const int lane = tid & 31, wave = tid >> 5;
  const int half = lane >> 4, m = lane & 15;

  const int tilesY = H >> 4;
  const int bw = ((W >> 4) + 7) >> 3;  // blocks across (8 tiles per block)
  const long blk = blockIdx.x;
  const int b = (int)(blk / ((long)tilesY * bw));
  const int rem = (int)(blk - (long)b * tilesY * bw);
  const int ty = rem / bw, bx = rem % bw;
  const int r0 = ty << 4, cb = bx << 7;
  const int startcol = (cb - 8) < 0 ? 0 : (cb - 8);
  const float* img = src + (long)b * H * W;
  const unsigned lds_base = (unsigned)(size_t)&smem[0];

  if (wave == 0) {  // one wave issues the TDM transfer (EXEC-independent op)
    const unsigned long long ga =
        (unsigned long long)(size_t)(img + (long)r0 * W + startcol);
    u32x4 g0;
    g0[0] = 1u;                                   // count=1 user descriptor
    g0[1] = lds_base;                             // lds_addr (bytes)
    g0[2] = (unsigned)(ga & 0xFFFFFFFFu);         // global_addr[31:0]
    g0[3] = (unsigned)((ga >> 32) & 0x01FFFFFFu) | (2u << 30);  // [56:32]|type=2
    const unsigned dim0 = (unsigned)(W - startcol);
    const unsigned dim1 = (unsigned)(H - r0);
    i32x8 g1;
    g1[0] = (int)((2u << 16) | (1u << 20) | (3u << 22));  // 4B elems, pad 1dw/16dw
    g1[1] = (int)((dim0 & 0xFFFFu) << 16);                // tensor_dim0 lo
    g1[2] = (int)((dim0 >> 16) | ((dim1 & 0xFFFFu) << 16));
    g1[3] = (int)((dim1 >> 16) | (144u << 16));           // tile_dim0 = 144
    g1[4] = 16;                                           // tile_dim1=16, dim2=0
    g1[5] = W;                                            // tensor_dim0_stride
    g1[6] = 0;
    g1[7] = 0;
    i32x4 gz = {0, 0, 0, 0};                              // 2D: groups 2/3 unused
    i32x8 gz8 = {0, 0, 0, 0, 0, 0, 0, 0};
    __builtin_amdgcn_tensor_load_to_lds(g0, g1, gz, gz, gz8, 0);
    __builtin_amdgcn_s_wait_tensorcnt(0);
  }
  __syncthreads();

  const int c0 = cb + (wave << 4);
  if (c0 >= W) return;  // tail waves; barrier already passed, EXEC all-1s below

  // A fragment from LDS: element e -> K = e + 16*half, col c0-8+K.
  v16h a;
#pragma unroll
  for (int e = 0; e < 16; ++e) {
    const int K = e + 16 * half;
    const int col = c0 - 8 + K;
    int lc = col - startcol;
    lc = lc < 0 ? 0 : lc;
    const float v = smem[m * 153 + lc + (lc >> 4)] *
                    ((col >= 0 && col < W) ? 1.0f : 0.0f);
    a[e] = (_Float16)v;
  }
  // B fragment: two 128-bit loads from the per-lane band table.
  const v16h bf = *(const v16h*)(bh + lane * 16);

  v8f c = {};
  c = __builtin_amdgcn_wmma_f32_16x16x32_f16(false, a, false, bf,
                                             (short)0, c, false, false);
  float* out = dst + (long)b * H * W;
#pragma unroll
  for (int v = 0; v < 8; ++v)  // D: element v -> M = v + 8*half, N = m
    out[(long)(r0 + v + 8 * half) * W + c0 + m] = c[v];
}

// ---------------------------------------------------------------------------
// Vertical pass (x100 folded into the band). Block = 8 waves = 8 adjacent
// 16x16 tiles. 32-row x 128-col halo tile staged via async-to-LDS (ASYNCcnt).
// ---------------------------------------------------------------------------
__global__ void smooth_v_k(const float* __restrict__ src, float* __restrict__ dst,
                           const _Float16* __restrict__ av, int B, int H, int W) {
  __shared__ float smem[32 * 132];  // 132-float pitch: conflict-free frag reads
  const int tid = threadIdx.x;
  const int lane = tid & 31, wave = tid >> 5;
  const int half = lane >> 4, n = lane & 15;

  const int tilesY = H >> 4;
  const int bw = ((W >> 4) + 7) >> 3;
  const long blk = blockIdx.x;
  const int b = (int)(blk / ((long)tilesY * bw));
  const int rem = (int)(blk - (long)b * tilesY * bw);
  const int ty = rem / bw, bx = rem % bw;
  const int r0 = ty << 4, cb = bx << 7;
  const float* img = src + (long)b * H * W;

  // Async stage: rows r0-8..r0+23 (clamped), cols cb..cb+127 (clamped).
  const unsigned lds_base = (unsigned)(size_t)&smem[0];
#pragma unroll
  for (int it = 0; it < 4; ++it) {
    const int q = tid + it * 256;          // 1024 x b128 chunks
    const int row = q >> 5;                // 0..31
    const int cq = (q & 31) << 2;          // 0..124 (floats)
    int rg = r0 - 8 + row;
    rg = rg < 0 ? 0 : (rg >= H ? H - 1 : rg);
    int cg = cb + cq;
    cg = cg > W - 4 ? W - 4 : cg;          // clamped lanes never read back
    const float* gp = img + (long)rg * W + cg;
    const unsigned la = lds_base + (unsigned)((row * 132 + cq) * 4);
    asm volatile("global_load_async_to_lds_b128 %0, %1, off"
                 :: "v"(la), "v"(gp) : "memory");
  }
  asm volatile("s_wait_asynccnt 0x0" ::: "memory");
  __syncthreads();

  const int c0 = cb + (wave << 4);
  if (c0 >= W) return;

  const v16h a = *(const v16h*)(av + lane * 16);  // band A fragment

  // B fragment from LDS: element e -> K = (e<8?e:e+8)+8*half, row r0-8+K, col c0+n.
  v16h bf;
#pragma unroll
  for (int e = 0; e < 16; ++e) {
    const int K = (e < 8 ? e : e + 8) + 8 * half;
    const int rg = r0 - 8 + K;
    const float v = (rg >= 0 && rg < H) ? smem[K * 132 + (wave << 4) + n] : 0.0f;
    bf[e] = (_Float16)v;
  }
  v8f c = {};
  c = __builtin_amdgcn_wmma_f32_16x16x32_f16(false, a, false, bf,
                                             (short)0, c, false, false);
  float* out = dst + (long)b * H * W;
#pragma unroll
  for (int v = 0; v < 8; ++v)
    out[(long)(r0 + v + 8 * half) * W + c0 + n] = c[v];
}

// ---------------------------------------------------------------------------
// Patch scoring: one wave per (b, patch). errsum is reused for the masked
// losses later (no second full-image pass).
// ---------------------------------------------------------------------------
__global__ void score_k(const float* __restrict__ outp, const float* __restrict__ tl,
                        float* __restrict__ score, float* __restrict__ errsum,
                        int B, int Hh, int k, int ires) {
  const int lane = threadIdx.x & 31;
  const long patch = (long)blockIdx.x * (blockDim.x >> 5) + (threadIdx.x >> 5);
  const int PP = 24 * 24;
  if (patch >= (long)B * PP) return;
  const int b = (int)(patch / PP);
  const int p = (int)(patch % PP);
  const int py = p / 24, px = p % 24;
  const float* o = outp + (long)b * Hh * Hh;
  const float* tt = tl + (long)b * Hh * Hh;
  __builtin_prefetch(o + (long)(py * k) * Hh + px * k, 0, 1);

  float es = 0.0f, ts = 0.0f;
  const int kk = k * k;
  for (int idx = lane; idx < kk; idx += 32) {
    const int dy = idx / k, dx = idx - dy * k;
    const long off = (long)(py * k + dy) * Hh + (px * k + dx);
    const float tv = tt[off];
    const float d = o[off] - tv;
    es += d * d;
    ts += tv;
  }
  for (int s = 16; s; s >>= 1) {  // wave32 reduction
    es += __shfl_xor(es, s, 32);
    ts += __shfl_xor(ts, s, 32);
  }
  if (lane == 0) {
    const long sidx = ((long)b * 4 + ires) * PP + p;
    errsum[sidx] = es;
    score[sidx] = es / (float)kk + es / (ts + 1e-10f);
  }
}

// ---------------------------------------------------------------------------
// Argmin routing + loss. Patch contributes errsum_i for all i <= moe
// (mask_add == [moe >= i]); loss_i normalized by B*H_i^2, weights all 1.
// ---------------------------------------------------------------------------
__global__ void moe_loss_k(const float* __restrict__ score, const float* __restrict__ errsum,
                           int* __restrict__ moe_i, float* __restrict__ moe_f,
                           float* __restrict__ loss, int B) {
  const int PP = 576;
  const long id = (long)blockIdx.x * blockDim.x + threadIdx.x;
  float lsum = 0.0f;
  if (id < (long)B * PP) {
    const int b = (int)(id / PP);
    const int p = (int)(id % PP);
    const float* s = score + (long)b * 4 * PP + p;
    float best = s[0];
    int bi = 0;
#pragma unroll
    for (int i = 1; i < 4; ++i) {  // strict < : first-occurrence argmin
      const float v = s[(long)i * PP];
      if (v < best) { best = v; bi = i; }
    }
    moe_i[id] = bi;
    moe_f[id] = (float)bi;
    const float* e = errsum + (long)b * 4 * PP + p;
    for (int i = 0; i <= bi; ++i) {
      const int Hi = 768 >> i;
      lsum += e[(long)i * PP] / ((float)B * (float)Hi * (float)Hi);
    }
  }
  for (int s = 16; s; s >>= 1) lsum += __shfl_xor(lsum, s, 32);
  if ((threadIdx.x & 31) == 0) atomicAdd(loss, lsum);
}

__global__ void zero_loss_k(float* loss) {
  if (threadIdx.x == 0 && blockIdx.x == 0) *loss = 0.0f;
}

// ---------------------------------------------------------------------------
// gt_den: per-pixel gather from tl_moe with centered 0.2 padding, / WEIGHT.
// ---------------------------------------------------------------------------
__global__ void gtden_k(const float* __restrict__ tl0, const float* __restrict__ tl1,
                        const float* __restrict__ tl2, const float* __restrict__ tl3,
                        const int* __restrict__ moe, float* __restrict__ gt_den, int B) {
  const long id = (long)blockIdx.x * blockDim.x + threadIdx.x;
  const long N = (long)B * 768 * 768;
  if (id >= N) return;
  const int b = (int)(id / (768 * 768));
  const int p = (int)(id % (768 * 768));
  const int y = p / 768, x = p % 768;
  const int py = y >> 5, px = x >> 5;
  const int m = moe[((long)b * 24 + py) * 24 + px];
  float val;
  if (m == 0) {
    val = tl0[id];
  } else {
    const int k = 32 >> m;
    const int pad = (32 - k) >> 1;
    const int iy = (y & 31) - pad, ix = (x & 31) - pad;
    if (iy >= 0 && iy < k && ix >= 0 && ix < k) {
      const int Hi = 768 >> m;
      const float* tl = (m == 1) ? tl1 : ((m == 2) ? tl2 : tl3);
      val = tl[((long)b * Hi + (py * k + iy)) * Hi + (px * k + ix)];
    } else {
      val = 0.2f;
    }
  }
  gt_den[id] = val * 0.01f;  // / WEIGHT
}

// ---------------------------------------------------------------------------
extern "C" void kernel_launch(void* const* d_in, const int* in_sizes, int n_in,
                              void* d_out, int out_size, void* d_ws, size_t ws_size,
                              hipStream_t stream) {
  const float* outp[4] = {(const float*)d_in[0], (const float*)d_in[1],
                          (const float*)d_in[2], (const float*)d_in[3]};
  const float* tgtp[4] = {(const float*)d_in[4], (const float*)d_in[5],
                          (const float*)d_in[6], (const float*)d_in[7]};
  const int B = in_sizes[0] / (768 * 768);

  // Workspace (floats): tl0..tl3 | tmp | score | errsum | moe(int) | band tables
  float* w = (float*)d_ws;
  size_t S[4], tl_off[4], off = 0;
  for (int i = 0; i < 4; ++i) {
    const int Hi = 768 >> i;
    S[i] = (size_t)B * Hi * Hi;
    tl_off[i] = off;
    off += S[i];
  }
  const size_t tmp_off = off;   off += S[0];
  const size_t score_off = off; off += (size_t)B * 4 * 576;
  const size_t err_off = off;   off += (size_t)B * 4 * 576;
  int* moe_ws = (int*)(w + off); off += (size_t)B * 576;
  _Float16* bh = (_Float16*)(w + off);   // 32 lanes x 16 f16
  _Float16* av = bh + 512;

  float* loss = (float*)d_out;           // d_out: [loss | gt_den | moe]
  float* gt_den = (float*)d_out + 1;
  float* moe_f = (float*)d_out + 1 + S[0];

  zero_loss_k<<<1, 64, 0, stream>>>(loss);
  band_init_k<<<1, 32, 0, stream>>>(bh, av);

  // Separable Gaussian smoothing via WMMA (8 tiles per 256-thread block).
  for (int i = 0; i < 4; ++i) {
    const int Hi = 768 >> i;
    const int bw = ((Hi >> 4) + 7) >> 3;          // blocks across
    const long nblk = (long)B * (Hi >> 4) * bw;
    smooth_h_k<<<(int)nblk, 256, 0, stream>>>(
        tgtp[i], w + tmp_off, bh, B, Hi, Hi);
    smooth_v_k<<<(int)nblk, 256, 0, stream>>>(
        w + tmp_off, w + tl_off[i], av, B, Hi, Hi);
  }

  // Patch scoring: one wave per (b, patch).
  {
    const long npatch = (long)B * 576;
    const int blocks = (int)((npatch + 7) / 8);
    for (int i = 0; i < 4; ++i) {
      const int Hi = 768 >> i;
      score_k<<<blocks, 256, 0, stream>>>(outp[i], w + tl_off[i], w + score_off,
                                          w + err_off, B, Hi, 32 >> i, i);
    }
  }

  // Routing + loss.
  {
    const long n = (long)B * 576;
    moe_loss_k<<<(int)((n + 255) / 256), 256, 0, stream>>>(
        w + score_off, w + err_off, moe_ws, moe_f, loss, B);
  }

  // gt_den assembly.
  {
    const long n = (long)B * 768 * 768;
    gtden_k<<<(int)((n + 255) / 256), 256, 0, stream>>>(
        w + tl_off[0], w + tl_off[1], w + tl_off[2], w + tl_off[3],
        moe_ws, gt_den, B);
  }
}
